// VQEmbedding_11450382811481
// MI455X (gfx1250) — matrix operand order repaired and checked by reading, not compiled, not run on hardware
//
#include <hip/hip_runtime.h>
#include <stdint.h>

// VQ nearest-codebook-index: N=32768 rows, D=256, K=8192 codes.
// dist = ||e||^2 - 2 z.e ; GEMM (N x K x D) on bf16 WMMA with hi/lo split.
// B tiles double-buffered in 128KB LDS (shared by 8 waves) via async g->LDS.

#define KCODES 8192
#define DDIM   256
#define NROWS  32768
#define NT     64                  // codes staged per iteration
#define NSTAGES (KCODES / NT)      // 128
#define ROWB   (NT * 2)            // LDS bytes per d-row (128)
#define NCHUNK (ROWB / 16)         // 16B chunks per row (8)

typedef __attribute__((ext_vector_type(16))) __bf16 v16bf;
typedef __attribute__((ext_vector_type(8)))  __bf16 v8bf;
typedef __attribute__((ext_vector_type(8)))  float  v8f;

// 16-byte payload type matching the async builtin's parameter type
typedef int b128_t __attribute__((vector_size(16)));

#if defined(__has_builtin)
#  if __has_builtin(__builtin_amdgcn_global_load_async_to_lds_b128)
#    define HAVE_ASYNC_LDS 1
#  endif
#  if __has_builtin(__builtin_amdgcn_s_wait_asynccnt)
#    define HAVE_WAIT_ASYNC 1
#  endif
#endif

#if defined(HAVE_ASYNC_LDS)
#  if defined(HAVE_WAIT_ASYNC)
#    define ASYNC_WAIT(n) __builtin_amdgcn_s_wait_asynccnt(n)
#  else
#    define ASYNC_WAIT(n) asm volatile("s_wait_asynccnt %0" ::"n"(n) : "memory")
#  endif
#else
#  define ASYNC_WAIT(n) do {} while (0)
#endif

static __device__ __forceinline__ void copy16_g2l(void* lds_dst, const void* gsrc) {
#if defined(HAVE_ASYNC_LDS)
  __builtin_amdgcn_global_load_async_to_lds_b128(
      (__attribute__((address_space(1))) b128_t*)(uintptr_t)gsrc,
      (__attribute__((address_space(3))) b128_t*)(uintptr_t)lds_dst,
      0, 0);
#else
  *(uint4*)lds_dst = *(const uint4*)gsrc;
#endif
}

static __device__ __forceinline__ unsigned short f32_to_bf16(float f) {
  unsigned u = __float_as_uint(f);
  u += 0x7FFFu + ((u >> 16) & 1u);          // round-to-nearest-even
  return (unsigned short)(u >> 16);
}
static __device__ __forceinline__ float bf16_hi_to_f32(unsigned short h) {
  return __uint_as_float(((unsigned)h) << 16);
}

// ---------------- prep 1: split z_e_x into bf16 hi/lo, row-major [N][D] ----
__global__ void vq_split_kernel(const float* __restrict__ x,
                                unsigned short* __restrict__ hi,
                                unsigned short* __restrict__ lo, int n) {
  int i = blockIdx.x * blockDim.x + threadIdx.x;
  if (i >= n) return;
  float v = x[i];
  unsigned short h = f32_to_bf16(v);
  hi[i] = h;
  lo[i] = f32_to_bf16(v - bf16_hi_to_f32(h));
}

// ---------------- prep 2: codebook -> transposed bf16 hi/lo [D][K] ---------
__global__ void vq_cbT_kernel(const float* __restrict__ cb,
                              unsigned short* __restrict__ tHi,
                              unsigned short* __restrict__ tLo) {
  int i = blockIdx.x * blockDim.x + threadIdx.x;   // over K*D, [k][d]
  int k = i >> 8;
  int d = i & 255;
  float v = cb[i];
  unsigned short h = f32_to_bf16(v);
  tHi[(size_t)d * KCODES + k] = h;
  tLo[(size_t)d * KCODES + k] = f32_to_bf16(v - bf16_hi_to_f32(h));
}

// ---------------- prep 3: cb_sq[k] = sum_d cb[k][d]^2 ----------------------
__global__ void vq_cbsq_kernel(const float* __restrict__ cb,
                               float* __restrict__ sq) {
  int k = blockIdx.x;
  float v = cb[(size_t)k * DDIM + threadIdx.x];
  float s = v * v;
  __shared__ float red[8];
  for (int o = 16; o >= 1; o >>= 1) s += __shfl_down(s, o, 32);
  if ((threadIdx.x & 31) == 0) red[threadIdx.x >> 5] = s;
  __syncthreads();
  if (threadIdx.x < 32) {
    float t = (threadIdx.x < 8) ? red[threadIdx.x] : 0.0f;
    for (int o = 4; o >= 1; o >>= 1) t += __shfl_down(t, o, 32);
    if (threadIdx.x == 0) sq[k] = t;
  }
}

// --- stage copy: thread t copies d-row t (NT codes, hi+lo) into LDS --------
// 128B rows stride 32 banks, so lanes alias with period 2 in row index;
// XOR the 16B-chunk index with (row>>1)&7 -> (row&1, chunk') injective over
// the 16 lanes of a fragment read -> conflict-free ds_load_b128.
static __device__ __forceinline__ void
stage_copy(unsigned short* __restrict__ sH, unsigned short* __restrict__ sL,
           const unsigned short* __restrict__ gH,
           const unsigned short* __restrict__ gL, int n0, int t) {
  const char* srcH = (const char*)(gH + (size_t)t * KCODES + n0);
  const char* srcL = (const char*)(gL + (size_t)t * KCODES + n0);
  char* dH = (char*)sH + t * ROWB;
  char* dL = (char*)sL + t * ROWB;
  const int swz = (t >> 1) & 7;
#pragma unroll
  for (int j = 0; j < NCHUNK; ++j) {
    int pj = 16 * (j ^ swz);
    copy16_g2l(dH + pj, srcH + j * 16);
    copy16_g2l(dL + pj, srcL + j * 16);
  }
}

// --- B fragment from LDS (ISA B 32x16 layout: lane holds N=lane%16, K in table order) ---
static __device__ __forceinline__ v16bf
load_bfrag(const unsigned short* __restrict__ s, int rowk, int sub) {
  const int kb = rowk & ~31;
  const int h  = (rowk & 31) >> 4;
  const int n  = sub * 16 + (rowk & 15);
  v16bf r;
#pragma unroll
  for (int e = 0; e < 16; ++e) {
    const int k = kb + (e < 8 ? 8 * h + e : 16 + 8 * h + (e - 8));
    const char* p = (const char*)s + k * ROWB + 16 * ((n >> 3) ^ ((k >> 1) & 7)) + (n & 7) * 2;
    r[e] = *(const __bf16*)p;
  }
  return r;
}

// ---------------- main: WMMA GEMM + fused argmin ---------------------------
// Block = 256 threads = 8 waves, 128 rows per WG, grid = 256.
// Double-buffered LDS staging of 64-code B tiles; 96 WMMAs per stage.
__global__ void __launch_bounds__(256)
vq_argmin_wmma_kernel(const unsigned short* __restrict__ zs_hi,
                      const unsigned short* __restrict__ zs_lo,
                      const unsigned short* __restrict__ cbT_hi,
                      const unsigned short* __restrict__ cbT_lo,
                      const float* __restrict__ cb_sq,
                      float* __restrict__ out) {
  __shared__ __align__(16) unsigned short sH[2][DDIM * NT];   // 2 x 32KB
  __shared__ __align__(16) unsigned short sL[2][DDIM * NT];   // 2 x 32KB

  const int t    = threadIdx.x;
  const int lane = t & 31;
  const int wave = t >> 5;
  const int R    = blockIdx.x * 128 + wave * 16;
  const int row  = lane & 15;
  const int half = lane >> 4;

  // A fragments (ISA 16-bit A 16x32 layout), resident in registers
  v16bf aH[8], aL[8];
#pragma unroll
  for (int c = 0; c < 8; ++c) {
    size_t base = (size_t)(R + row) * DDIM + (size_t)(c * 32 + half * 8);
    v8bf h0 = *(const v8bf*)(zs_hi + base);
    v8bf h1 = *(const v8bf*)(zs_hi + base + 16);
    v8bf l0 = *(const v8bf*)(zs_lo + base);
    v8bf l1 = *(const v8bf*)(zs_lo + base + 16);
    aH[c] = __builtin_shufflevector(h0, h1, 0,1,2,3,4,5,6,7,8,9,10,11,12,13,14,15);
    aL[c] = __builtin_shufflevector(l0, l1, 0,1,2,3,4,5,6,7,8,9,10,11,12,13,14,15);
  }

  float minv[8];
  int   mini[8];
#pragma unroll
  for (int i = 0; i < 8; ++i) { minv[i] = 3.402823466e38f; mini[i] = 0; }

  // prologue: stage 0 into buffer 0 (16 async b128 per thread)
  stage_copy(sH[0], sL[0], cbT_hi, cbT_lo, 0, t);

  for (int it = 0; it < NSTAGES; ++it) {
    const int cur = it & 1;
    __syncthreads();   // everyone done computing on the buffer we overwrite next
    if (it + 1 < NSTAGES) {
      stage_copy(sH[cur ^ 1], sL[cur ^ 1], cbT_hi, cbT_lo, (it + 1) * NT, t);
      ASYNC_WAIT(16);  // async completes in-order: current stage's 16 are done
    } else {
      ASYNC_WAIT(0);
    }
    __syncthreads();   // all waves' slices of current stage resident

#pragma unroll
    for (int sub = 0; sub < NT / 16; ++sub) {
      const int n0 = it * NT + sub * 16;
      v8f acc = {};
#pragma unroll
      for (int c = 0; c < 8; ++c) {
        const int rk = c * 32 + lane;
        v16bf bH = load_bfrag(sH[cur], rk, sub);
        v16bf bL = load_bfrag(sL[cur], rk, sub);
        acc = __builtin_amdgcn_wmma_f32_16x16x32_bf16(false, aH[c], false, bH,
                                                      (short)0, acc, false, false);
        acc = __builtin_amdgcn_wmma_f32_16x16x32_bf16(false, aH[c], false, bL,
                                                      (short)0, acc, false, false);
        acc = __builtin_amdgcn_wmma_f32_16x16x32_bf16(false, aL[c], false, bH,
                                                      (short)0, acc, false, false);
      }
      // C layout: acc[i] is row (i + 8*half), col = lane&15 (+n0)
      float csq = cb_sq[n0 + row];
      int   col = n0 + row;
#pragma unroll
      for (int i = 0; i < 8; ++i) {
        float dst = __builtin_fmaf(-2.0f, acc[i], csq);
        if (dst < minv[i]) { minv[i] = dst; mini[i] = col; }
      }
    }
  }

  // Cross-lane argmin over the 16 lanes holding the same row
#pragma unroll
  for (int i = 0; i < 8; ++i) {
    float v  = minv[i];
    int   id = mini[i];
#pragma unroll
    for (int s = 1; s < 16; s <<= 1) {
      float ov = __shfl_xor(v, s, 32);
      int   oi = __shfl_xor(id, s, 32);
      if (ov < v || (ov == v && oi < id)) { v = ov; id = oi; }
    }
    if (lane == 0)  out[R + i]     = (float)id;
    if (lane == 16) out[R + 8 + i] = (float)id;
  }
}

// ---------------------------------------------------------------------------
extern "C" void kernel_launch(void* const* d_in, const int* in_sizes, int n_in,
                              void* d_out, int out_size, void* d_ws, size_t ws_size,
                              hipStream_t stream) {
  (void)in_sizes; (void)n_in; (void)out_size; (void)ws_size;
  const float* z  = (const float*)d_in[0];   // [32,32,32,256] f32
  const float* cb = (const float*)d_in[1];   // [8192,256]    f32
  float* out = (float*)d_out;                // [32768] indices as f32

  // workspace carve-out (~42 MB)
  unsigned short* zs_hi  = (unsigned short*)d_ws;                  // N*D bf16
  unsigned short* zs_lo  = zs_hi  + (size_t)NROWS * DDIM;
  unsigned short* cbT_hi = zs_lo  + (size_t)NROWS * DDIM;          // [D][K]
  unsigned short* cbT_lo = cbT_hi + (size_t)KCODES * DDIM;
  float*          cb_sq  = (float*)(cbT_lo + (size_t)KCODES * DDIM);

  vq_split_kernel<<<(NROWS * DDIM) / 256, 256, 0, stream>>>(z, zs_hi, zs_lo,
                                                            NROWS * DDIM);
  vq_cbT_kernel<<<(KCODES * DDIM) / 256, 256, 0, stream>>>(cb, cbT_hi, cbT_lo);
  vq_cbsq_kernel<<<KCODES, 256, 0, stream>>>(cb, cb_sq);
  vq_argmin_wmma_kernel<<<NROWS / 128, 256, 0, stream>>>(zs_hi, zs_lo,
                                                         cbT_hi, cbT_lo,
                                                         cb_sq, out);
}
